// LorentzAtt_35639638622838
// MI455X (gfx1250) — matrix operand order repaired and checked by reading, not compiled, not run on hardware
//
#include <hip/hip_runtime.h>
#include <hip/hip_bf16.h>

// ---------------------------------------------------------------------------
// Lorentz attention, MI455X (gfx1250, wave32, WMMA).
//   N = 8192 rows, D = 256 features.
//   HBM-bound on the 256MB att_adj matrix (write once, read once);
//   GEMMs run on v_wmma_f32_16x16x32_bf16 with f32 accumulation.
// ---------------------------------------------------------------------------

#define NROWS 8192
#define DDIM  256

typedef __attribute__((ext_vector_type(16))) __bf16 v16bf;
typedef __attribute__((ext_vector_type(8)))  float  v8f;

__device__ __forceinline__ int lane_id() { return (int)(threadIdx.x & 31); }

// A-fragment (16Mx32K, bf16, row-major source, leading dim ld).
// Layout per ISA 7.12.2: lane half h, row m=lane&15; elements 0..7 -> K = k0+8h+e,
// elements 8..15 -> K = k0+16+8h+e  (two contiguous 16B runs -> b128 loads).
__device__ __forceinline__ v16bf load_fragA_bf16(const __bf16* __restrict__ base,
                                                 int ld, int row0, int k0) {
  int lane = lane_id();
  int half = lane >> 4;
  int m    = lane & 15;
  const __bf16* p = base + (size_t)(row0 + m) * ld + (k0 + half * 8);
  v16bf a;
#pragma unroll
  for (int e = 0; e < 8; ++e) a[e]     = p[e];
#pragma unroll
  for (int e = 0; e < 8; ++e) a[8 + e] = p[16 + e];
  return a;
}

// Same A layout but from f32 source, converting to bf16 in-register.
__device__ __forceinline__ v16bf load_fragA_f32(const float* __restrict__ base,
                                                int ld, int row0, int k0) {
  int lane = lane_id();
  int half = lane >> 4;
  int m    = lane & 15;
  const float* p = base + (size_t)(row0 + m) * ld + (k0 + half * 8);
  v16bf a;
#pragma unroll
  for (int e = 0; e < 8; ++e) a[e]     = (__bf16)p[e];
#pragma unroll
  for (int e = 0; e < 8; ++e) a[8 + e] = (__bf16)p[16 + e];
  return a;
}

// B-fragment (32Kx16N, bf16). Source rows hold the N dimension (i.e. we load
// B[k][n] from Rows[n][k], which is exactly K^T / x^T). Layout per ISA 7.12.4
// analog: lane half h, column n=lane&15; elements 0..7 -> K = k0+8h+e,
// elements 8..15 -> K = k0+16+8h+e (two contiguous 16B runs -> b128 loads).
__device__ __forceinline__ v16bf load_fragB_bf16(const __bf16* __restrict__ base,
                                                 int ld, int n0, int k0) {
  int lane = lane_id();
  int half = lane >> 4;
  int n    = lane & 15;
  const __bf16* p = base + (size_t)(n0 + n) * ld + (k0 + half * 8);
  v16bf b;
#pragma unroll
  for (int e = 0; e < 8; ++e) b[e] = p[e];
#pragma unroll
  for (int e = 0; e < 8; ++e) b[8 + e] = p[16 + e];
  return b;
}

// ---------------------------------------------------------------------------
// Kernel 1: Lorentz linear. One block (256 threads) per row.
//   y = x[r] @ W^T + b ; time = sigmoid(y0)*exp(log_scale)+1.1
//   out = [(+/-)time, spatial*sqrt((time^2-1)/clip(|spatial|^2,1e-8))] as bf16.
// negate_time=1 for Q: folds the -2*q0*k0 Minkowski correction into the GEMM.
// ---------------------------------------------------------------------------
__global__ __launch_bounds__(256)
void lorentz_kernel(const float* __restrict__ x, const float* __restrict__ W,
                    const float* __restrict__ bias, const float* __restrict__ log_scale,
                    __bf16* __restrict__ outbf, int negate_time) {
  __shared__ float xrow[DDIM];
  __shared__ float red[DDIM];
  __shared__ float y0s;

  int r = blockIdx.x;
  int c = threadIdx.x;

  xrow[c] = x[(size_t)r * DDIM + c];
  __syncthreads();

  const float* wrow = W + (size_t)c * DDIM;
  float acc = 0.0f;
#pragma unroll 8
  for (int k = 0; k < DDIM; ++k) acc = fmaf(xrow[k], wrow[k], acc);
  acc += bias[c];

  red[c] = (c == 0) ? 0.0f : acc * acc;
  if (c == 0) y0s = acc;
  __syncthreads();
#pragma unroll
  for (int s = DDIM / 2; s > 0; s >>= 1) {
    if (c < s) red[c] += red[c + s];
    __syncthreads();
  }

  float sq    = fmaxf(red[0], 1e-8f);
  float t     = 1.0f / (1.0f + __expf(-y0s)) * __expf(log_scale[0]) + 1.1f;
  float scale = (t * t - 1.0f) / sq;
  float sr    = sqrtf(scale);

  float val = (c == 0) ? (negate_time ? -t : t) : acc * sr;
  outbf[(size_t)r * DDIM + c] = (__bf16)val;
}

// ---------------------------------------------------------------------------
// Kernel 2: x (N x D f32) -> x^T (D x N bf16) for contiguous B-fragments.
// ---------------------------------------------------------------------------
__global__ __launch_bounds__(256)
void xT_bf16_kernel(const float* __restrict__ x, __bf16* __restrict__ xT) {
  size_t idx = (size_t)blockIdx.x * 256 + threadIdx.x;   // over N*D
  int r = (int)(idx >> 8);
  int c = (int)(idx & 255);
  xT[(size_t)c * NROWS + r] = (__bf16)x[idx];
}

// ---------------------------------------------------------------------------
// Kernel 3: att_adj = sigmoid((2 + 2*(Qn @ Kb^T))/att_scale + att_bias).
// Block tile 64x32 (8 waves, one 16x16 WMMA tile each), K=256 unrolled.
// Q/K operands are 4MB bf16 -> L2 resident; the 256MB f32 store dominates.
// ---------------------------------------------------------------------------
__global__ __launch_bounds__(256)
void att_kernel(const __bf16* __restrict__ Qb, const __bf16* __restrict__ Kb,
                float* __restrict__ att,
                const float* __restrict__ att_bias_p,
                const float* __restrict__ att_scale_p) {
  int w    = threadIdx.x >> 5;          // wave 0..7
  int wm   = w & 3, wn = w >> 2;        // 4x2 wave grid
  int row0 = blockIdx.y * 64 + wm * 16;
  int col0 = blockIdx.x * 32 + wn * 16;

  v8f acc = {};
#pragma unroll
  for (int k0 = 0; k0 < DDIM; k0 += 32) {
    v16bf a = load_fragA_bf16(Qb, DDIM, row0, k0);
    v16bf b = load_fragB_bf16(Kb, DDIM, col0, k0);
    acc = __builtin_amdgcn_wmma_f32_16x16x32_bf16(false, a, false, b,
                                                  (short)0, acc, false, false);
  }

  float inv_s = 2.0f / att_scale_p[0];
  float z0    = att_bias_p[0] + inv_s;  // (2+2c)/s + bias = c*(2/s) + (bias + 2/s)

  int lane = lane_id();
  int half = lane >> 4;
  int n    = lane & 15;
#pragma unroll
  for (int rr = 0; rr < 8; ++rr) {      // D layout: vgpr rr -> M = rr + 8*half
    int m   = rr + half * 8;
    float z = acc[rr] * inv_s + z0;
    att[(size_t)(row0 + m) * NROWS + (col0 + n)] = 1.0f / (1.0f + __expf(-z));
  }
}

// ---------------------------------------------------------------------------
// Kernel 4: support = att @ x   (M=8192, N=256, K=8192).
// Each workgroup owns 16 rows and ALL 256 columns, so the 256MB att matrix
// is streamed from HBM exactly once (single-pass read -> roofline floor).
// Wave w computes the 16x32 column slab [w*32, w*32+32): one shared A-frag
// (same addresses across all 8 waves -> WGP$/L2 hits) + two B-frags from the
// L2-resident bf16 x^T, two WMMAs per K-chunk. ~50 VGPRs -> high occupancy.
// ---------------------------------------------------------------------------
__global__ __launch_bounds__(256)
void support_kernel(const float* __restrict__ att, const __bf16* __restrict__ xT,
                    float* __restrict__ out) {
  int w    = threadIdx.x >> 5;          // wave 0..7 -> column slab
  int row0 = blockIdx.x * 16;
  int n0   = w * 32;

  v8f acc0 = {};
  v8f acc1 = {};
  for (int k0 = 0; k0 < NROWS; k0 += 32) {
    // pull row row0's att data 8 K-chunks ahead toward the caches while this one computes
    __builtin_prefetch(att + (size_t)row0 * NROWS + k0 + 256, 0, 1);
    v16bf a  = load_fragA_f32(att, NROWS, row0, k0);
    v16bf b0 = load_fragB_bf16(xT, NROWS, n0,      k0);
    v16bf b1 = load_fragB_bf16(xT, NROWS, n0 + 16, k0);
    acc0 = __builtin_amdgcn_wmma_f32_16x16x32_bf16(false, a, false, b0,
                                                   (short)0, acc0, false, false);
    acc1 = __builtin_amdgcn_wmma_f32_16x16x32_bf16(false, a, false, b1,
                                                   (short)0, acc1, false, false);
  }

  int lane = lane_id();
  int half = lane >> 4;
  int n    = lane & 15;
#pragma unroll
  for (int rr = 0; rr < 8; ++rr) {
    int m = rr + half * 8;
    out[(size_t)(row0 + m) * DDIM + (n0 + n)]      = acc0[rr];
    out[(size_t)(row0 + m) * DDIM + (n0 + 16 + n)] = acc1[rr];
  }
}

// ---------------------------------------------------------------------------
// Kernel 5: Lorentz normalization, in place on the output rows.
// ---------------------------------------------------------------------------
__global__ __launch_bounds__(256)
void normalize_kernel(float* __restrict__ out) {
  __shared__ float red[DDIM];
  __shared__ float s0s;

  int r = blockIdx.x;
  int c = threadIdx.x;
  float s = out[(size_t)r * DDIM + c];

  red[c] = (c == 0) ? 0.0f : s * s;
  if (c == 0) s0s = s;
  __syncthreads();
#pragma unroll
  for (int k = DDIM / 2; k > 0; k >>= 1) {
    if (c < k) red[c] += red[c + k];
    __syncthreads();
  }

  float neg_inner = s0s * s0s - red[0];
  float denorm    = sqrtf(fmaxf(fabsf(neg_inner), 1e-8f));
  out[(size_t)r * DDIM + c] = s / denorm;
}

// ---------------------------------------------------------------------------
// Launch. Inputs: x, Wq, bq, sq_log, Wk, bk, sk_log, att_bias, att_scale.
// d_out = [output (8192*256) | att_adj (8192*8192)] f32.
// d_ws  = [Qb 4MB | Kb 4MB | xT 4MB] bf16.
// ---------------------------------------------------------------------------
extern "C" void kernel_launch(void* const* d_in, const int* in_sizes, int n_in,
                              void* d_out, int out_size, void* d_ws, size_t ws_size,
                              hipStream_t stream) {
  const float* x       = (const float*)d_in[0];
  const float* Wq      = (const float*)d_in[1];
  const float* bq      = (const float*)d_in[2];
  const float* sq_log  = (const float*)d_in[3];
  const float* Wk      = (const float*)d_in[4];
  const float* bk      = (const float*)d_in[5];
  const float* sk_log  = (const float*)d_in[6];
  const float* att_b   = (const float*)d_in[7];
  const float* att_s   = (const float*)d_in[8];

  float* out = (float*)d_out;                        // 8192*256
  float* att = (float*)d_out + (size_t)NROWS * DDIM; // 8192*8192

  const size_t mat_bf16 = (size_t)NROWS * DDIM;      // elements
  __bf16* Qb = (__bf16*)d_ws;
  __bf16* Kb = Qb + mat_bf16;
  __bf16* xT = Kb + mat_bf16;

  // 1) Lorentz linears (Q gets negated time column -> Minkowski GEMM).
  lorentz_kernel<<<NROWS, DDIM, 0, stream>>>(x, Wq, bq, sq_log, Qb, 1);
  lorentz_kernel<<<NROWS, DDIM, 0, stream>>>(x, Wk, bk, sk_log, Kb, 0);

  // 2) x^T in bf16.
  xT_bf16_kernel<<<NROWS, DDIM, 0, stream>>>(x, xT);

  // 3) att_adj (WMMA bf16, fused sigmoid).  grid: 256 col-tiles x 128 row-tiles.
  att_kernel<<<dim3(NROWS / 32, NROWS / 64), 256, 0, stream>>>(Qb, Kb, att,
                                                               att_b, att_s);

  // 4) support = att @ x.  grid: 512 row blocks, each covers all 256 cols.
  support_kernel<<<NROWS / 16, 256, 0, stream>>>(att, xT, out);

  // 5) Lorentz normalize in place.
  normalize_kernel<<<NROWS, DDIM, 0, stream>>>(out);
}
